// ScaledDotProductAttention_3384434229740
// MI455X (gfx1250) — matrix operand order repaired and checked by reading, not compiled, not run on hardware
//
#include <hip/hip_runtime.h>

typedef __attribute__((ext_vector_type(16))) _Float16 v16h;
typedef __attribute__((ext_vector_type(8)))  _Float16 v8h;
typedef __attribute__((ext_vector_type(8)))  float    v8f;

constexpr int B_ = 2, H_ = 16, S_ = 2048, D_ = 128;
constexpr int KB   = 64;    // keys per block iteration (amortizes softmax overhead)
constexpr int KSTR = 136;   // halfs per K row in LDS (128 + 8 pad, breaks bank conflicts)
constexpr int VSTR = 72;    // halfs per V^T row in LDS (64 + 8 pad)
constexpr int PSTR = 72;    // halfs per P row in LDS  (64 + 8 pad)

// Softmax in log2 domain: v_exp_f32 is natively exp2, so fold log2(e) into the
// Q pre-scale and the mask fill value. p = 2^(s - m) == e^(logit/T - max/T).
constexpr float QSCALE2 = 0.08838834764831845f * 1.4426950408889634f; // log2(e)/sqrt(128)
#define NEG_INF2_ (-1.4426950408889634e9f)                            // -1e9 * log2(e)

__launch_bounds__(256, 1)
__global__ void fa_fwd(const float* __restrict__ q,
                       const float* __restrict__ kmat,
                       const float* __restrict__ vmat,
                       const int*   __restrict__ mask,
                       float* __restrict__ out)
{
  __shared__ _Float16 ks[KB * KSTR];      // K block, row-major f16      (~17 KB)
  __shared__ _Float16 vs[D_ * VSTR];      // V block, transposed f16     (~18 KB)
  __shared__ _Float16 ps[8 * 16 * PSTR];  // per-wave P relayout scratch (~18 KB)

  const int tid  = threadIdx.x;
  const int wave = tid >> 5;
  const int lane = tid & 31;
  const int lh   = lane & 15;   // lane within 16-lane half
  const int g    = lane >> 4;   // lane-half group (0/1)

  const int gid   = blockIdx.x;
  const int bh    = gid >> 4;          // b*H + h
  const int qtile = gid & 15;
  const int b     = bh >> 4;           // H == 16
  const int qbase = qtile * 128 + wave * 16;

  const float* qptr = q    + (size_t)bh * S_ * D_;
  const float* kptr = kmat + (size_t)bh * S_ * D_;
  const float* vptr = vmat + (size_t)bh * S_ * D_;
  const int*   mptr = mask + (size_t)b * S_ * S_;
  _Float16* psw = &ps[wave * 16 * PSTR];

  // ---- load Q tile (16 x 128), pre-scaled, into WMMA A-operand layout ----
  // 16-bit A 16x32: lane = M%16; lanes<16: e0..7 -> K0..7, e8..15 -> K16..23;
  //                 lanes>=16:   e0..7 -> K8..15, e8..15 -> K24..31.
  v16h aq[4];
  {
    const float* qrow = qptr + (size_t)(qbase + lh) * D_;
    #pragma unroll
    for (int t = 0; t < 4; ++t) {
      v8h lo, hi;
      const float* p0 = qrow + 32 * t + g * 8;
      const float* p1 = qrow + 32 * t + 16 + g * 8;
      #pragma unroll
      for (int i = 0; i < 8; ++i) {
        lo[i] = (_Float16)(p0[i] * QSCALE2);
        hi[i] = (_Float16)(p1[i] * QSCALE2);
      }
      ((v8h*)&aq[t])[0] = lo;
      ((v8h*)&aq[t])[1] = hi;
    }
  }

  v8f zero = {};
  v8f o[8];                       // output accumulators: 8 C tiles covering D=128
  #pragma unroll
  for (int dt = 0; dt < 8; ++dt) o[dt] = zero;
  float rmax[8], rsum[8];         // per-row softmax stats (C-layout row mapping)
  #pragma unroll
  for (int r = 0; r < 8; ++r) { rmax[r] = -1.0e30f; rsum[r] = 0.0f; }

  #pragma unroll 1
  for (int t0 = 0; t0 < S_; t0 += KB) {
    __syncthreads();   // protect ks/vs overwrite vs previous iteration's reads
    // ---- cooperative staging: K row-major f16, V transposed f16 ----
    {
      const int kr = tid >> 2;          // 0..63 key row
      const int kc = (tid & 3) << 5;    // 0,32,64,96 col base
      const float* kp = kptr + (size_t)(t0 + kr) * D_ + kc;
      const float* vp = vptr + (size_t)(t0 + kr) * D_ + kc;
      _Float16* kd = &ks[kr * KSTR + kc];
      #pragma unroll
      for (int j = 0; j < 32; j += 4) {
        float4 kf = *(const float4*)(kp + j);
        float4 vf = *(const float4*)(vp + j);
        kd[j + 0] = (_Float16)kf.x; kd[j + 1] = (_Float16)kf.y;
        kd[j + 2] = (_Float16)kf.z; kd[j + 3] = (_Float16)kf.w;
        vs[(kc + j + 0) * VSTR + kr] = (_Float16)vf.x;
        vs[(kc + j + 1) * VSTR + kr] = (_Float16)vf.y;
        vs[(kc + j + 2) * VSTR + kr] = (_Float16)vf.z;
        vs[(kc + j + 3) * VSTR + kr] = (_Float16)vf.w;
      }
    }
    __syncthreads();

    // ---- QK^T: four 16x16 score tiles, K-dim = 128 (4 WMMAs each) ----
    // 16-bit B 32x16: lane N = lane%16; e0..7 -> K = (lane/16)*8 + e, e8..15 -> K = 16 + (lane/16)*8 + (e-8).
    v8f s[4];
    #pragma unroll
    for (int nt = 0; nt < 4; ++nt) {
      v8f acc = zero;
      const _Float16* kb = &ks[(nt * 16 + lh) * KSTR + g * 8];
      #pragma unroll
      for (int t = 0; t < 4; ++t) {
        v16h bmat;
        ((v8h*)&bmat)[0] = *(const v8h*)(kb + 32 * t);
        ((v8h*)&bmat)[1] = *(const v8h*)(kb + 32 * t + 16);
        acc = __builtin_amdgcn_wmma_f32_16x16x32_f16(false, aq[t], false, bmat,
                                                     (short)0, acc, false, false);
      }
      s[nt] = acc;
    }

    // ---- apply mask in C layout (lane = key col, VGPR r = query row) ----
    #pragma unroll
    for (int nt = 0; nt < 4; ++nt) {
      const int tg = t0 + nt * 16 + lh;
      #pragma unroll
      for (int r = 0; r < 8; ++r) {
        const int mg = qbase + r + g * 8;
        const int mv = mptr[(size_t)mg * S_ + tg];
        s[nt][r] = mv ? s[nt][r] : NEG_INF2_;
      }
    }

    // ---- online softmax (base-2): row max/sum via 16-lane shfl_xor ----
    #pragma unroll
    for (int r = 0; r < 8; ++r) {
      float bmax = fmaxf(fmaxf(s[0][r], s[1][r]), fmaxf(s[2][r], s[3][r]));
      #pragma unroll
      for (int off = 8; off >= 1; off >>= 1)
        bmax = fmaxf(bmax, __shfl_xor(bmax, off, 32));
      const float nm   = fmaxf(rmax[r], bmax);
      const float corr = exp2f(rmax[r] - nm);
      float p[4];
      float bsum = 0.0f;
      #pragma unroll
      for (int nt = 0; nt < 4; ++nt) {
        p[nt] = exp2f(s[nt][r] - nm);
        bsum += p[nt];
      }
      #pragma unroll
      for (int off = 8; off >= 1; off >>= 1)
        bsum += __shfl_xor(bsum, off, 32);
      rmax[r] = nm;
      rsum[r] = rsum[r] * corr + bsum;
      #pragma unroll
      for (int dt = 0; dt < 8; ++dt) o[dt][r] *= corr;
      // scatter P into per-wave LDS scratch as (row, key) for A-layout reload
      #pragma unroll
      for (int nt = 0; nt < 4; ++nt)
        psw[(r + g * 8) * PSTR + nt * 16 + lh] = (_Float16)p[nt];
    }

    // ---- reload P as two A operands (keys 0..31 / 32..63), PV accumulate ----
    v16h pa[2];
    {
      const _Float16* pr = psw + lh * PSTR;
      #pragma unroll
      for (int c = 0; c < 2; ++c) {
        ((v8h*)&pa[c])[0] = *(const v8h*)(pr + c * 32 + g * 8);
        ((v8h*)&pa[c])[1] = *(const v8h*)(pr + c * 32 + 16 + g * 8);
      }
    }
    #pragma unroll
    for (int dt = 0; dt < 8; ++dt) {
      const _Float16* vb = &vs[(dt * 16 + lh) * VSTR + g * 8];
      #pragma unroll
      for (int c = 0; c < 2; ++c) {
        v16h bv;
        ((v8h*)&bv)[0] = *(const v8h*)(vb + c * 32);
        ((v8h*)&bv)[1] = *(const v8h*)(vb + c * 32 + 16);
        o[dt] = __builtin_amdgcn_wmma_f32_16x16x32_f16(false, pa[c], false, bv,
                                                       (short)0, o[dt], false, false);
      }
    }
  }

  // ---- normalize by row sums and store fp32 output ----
  float* obase = out + ((size_t)bh * S_ + qbase) * D_;
  #pragma unroll
  for (int r = 0; r < 8; ++r) {
    const float inv = 1.0f / rsum[r];
    float* op = obase + (size_t)(r + g * 8) * D_ + lh;
    #pragma unroll
    for (int dt = 0; dt < 8; ++dt)
      op[dt * 16] = o[dt][r] * inv;
  }
}

extern "C" void kernel_launch(void* const* d_in, const int* in_sizes, int n_in,
                              void* d_out, int out_size, void* d_ws, size_t ws_size,
                              hipStream_t stream) {
  // setup_inputs order: v, k, q, mask
  const float* v    = (const float*)d_in[0];
  const float* k    = (const float*)d_in[1];
  const float* q    = (const float*)d_in[2];
  const int*   mask = (const int*)d_in[3];
  float* out = (float*)d_out;

  const int nblocks = B_ * H_ * (S_ / 128);  // 512 workgroups, 8 waves each
  fa_fwd<<<nblocks, 256, 0, stream>>>(q, k, v, mask, out);
}
